// Vulnerability_backbone_50886772523452
// MI455X (gfx1250) — matrix operand order repaired and checked by reading, not compiled, not run on hardware
//
#include <hip/hip_runtime.h>
#include <hip/hip_bf16.h>
#include <stdint.h>

// ---------------- types ----------------
typedef __bf16 bf16_t;
typedef __attribute__((ext_vector_type(16))) __bf16 v16bf;
typedef __attribute__((ext_vector_type(8)))  __bf16 v8bf;
typedef __attribute__((ext_vector_type(8)))  float  v8f;

#define HDIM 256
#define FDIM 128
#define LDIM 8
#define TDIM 16
#define G3H  768   // 3*H

// CDNA5 async global->LDS copy (ASYNCcnt-tracked), with portable fallback.
// Probe-confirmed signature: (int4 AS1*, int4 AS3*, imm offset, imm cpol).
#if defined(__AMDGCN__) && __has_builtin(__builtin_amdgcn_global_load_async_to_lds_b128)
#define HAVE_ASYNC_LDS 1
typedef __attribute__((ext_vector_type(4))) int i4_t;
typedef __attribute__((address_space(1))) i4_t* gas_i4p;
typedef __attribute__((address_space(3))) i4_t* las_i4p;
#else
#define HAVE_ASYNC_LDS 0
#endif

// ---------------- device helpers ----------------
__device__ __forceinline__ bf16_t f2bf(float f) {
  union { float f; uint32_t u; } in; in.f = f;
  uint32_t u = in.u + 0x7FFFu + ((in.u >> 16) & 1u);   // round-to-nearest-even
  union { uint16_t s; bf16_t b; } out; out.s = (uint16_t)(u >> 16);
  return out.b;
}
__device__ __forceinline__ float sigf(float x) { return 1.0f / (1.0f + expf(-x)); }

// ---------------- WMMA GEMM: C[M,Ntot] = A[M,K] * W[Ntot,K]^T (+bias) ----------------
// Block = 8 waves = 128 M-rows x 64 N-cols. The 64xK W tile is shared by all 8
// waves, so it is staged through LDS (double-buffered 64x64-element chunks,
// async global->LDS when available). A fragments stream from global per wave.
// Fragment layouts per CDNA5 ISA 7.12.2 (wave32):
//   A 16x32 bf16 : lane<16 -> M=lane, K e<8:+e, e>=8:+8+e ; lane>=16 -> K base+8
//   B 32x16 bf16 : lane<16 -> N=lane, K e<8:+e, e>=8:+8+e ; lane>=16 -> N=lane-16, K base+8
//   C 16x16 f32  : lane<16 -> N=lane, M=r ; lane>=16 -> N=lane-16, M=8+r
#define WPITCH 72   // 64 data elems + 8 pad (16B) -> 144B pitch, conflict-free
__global__ __launch_bounds__(256) void k_gemm_bf16(
    const bf16_t* __restrict__ A, int lda,
    const bf16_t* __restrict__ W, int ldw,
    float* __restrict__ C, int ldc,
    const float* __restrict__ bias,
    int M, int Ntot, int K)
{
  __shared__ __align__(16) bf16_t sW[2][64 * WPITCH];

  const int tid   = threadIdx.x;
  const int lane  = tid & 31;
  const int wave  = tid >> 5;
  const int tileM = blockIdx.x * 8 + wave;
  const int n0    = blockIdx.y * 64;
  const int half  = lane >> 4;       // 0 or 1
  const int l15   = lane & 15;

  int mrow   = tileM * 16 + l15;
  int mclamp = (mrow < M) ? mrow : (M - 1);   // keep EXEC all-1s; mask at store
  const bf16_t* arow = A + (size_t)mclamp * lda + half * 8;

  // cooperative W loader: 512 x 16B per chunk; 2 consecutive 16B per thread
  const int wrow = tid >> 2;             // 0..63
  const int wcol = ((tid << 1) & 7) * 8; // elem col: 0,16,32,48
  const bf16_t* wsrc = W + (size_t)(n0 + wrow) * ldw + wcol;
  bf16_t* wdst0 = &sW[0][wrow * WPITCH + wcol];
  bf16_t* wdst1 = &sW[1][wrow * WPITCH + wcol];

  auto issue_chunk = [&](int c) {
    const bf16_t* g = wsrc + c * 64;
    bf16_t* s = (c & 1) ? wdst1 : wdst0;
#if HAVE_ASYNC_LDS
    __builtin_amdgcn_global_load_async_to_lds_b128(
        (gas_i4p)(void*)(g),     (las_i4p)(void*)(s),     0, 0);
    __builtin_amdgcn_global_load_async_to_lds_b128(
        (gas_i4p)(void*)(g + 8), (las_i4p)(void*)(s + 8), 0, 0);
#else
    *(v8bf*)s       = *(const v8bf*)g;
    *(v8bf*)(s + 8) = *(const v8bf*)(g + 8);
#endif
  };

  v8f acc[4] = {};
  const int nch = K >> 6;   // K is a multiple of 64 for every GEMM in this net
  issue_chunk(0);
  for (int c = 0; c < nch; ++c) {
    const bool more = (c + 1) < nch;
    if (more) issue_chunk(c + 1);
#if HAVE_ASYNC_LDS
    if (more) asm volatile("s_wait_asynccnt 0x2" ::: "memory");  // chunk c landed
    else      asm volatile("s_wait_asynccnt 0x0" ::: "memory");
#endif
    __syncthreads();                       // all waves' writes of buf[c&1] visible
    const bf16_t* sb = sW[c & 1];
#pragma unroll
    for (int kk = 0; kk < 64; kk += 32) {
      union { v16bf v; v8bf h[2]; } a;
      a.h[0] = *(const v8bf*)(arow + c * 64 + kk);        // K = +kA .. +7
      a.h[1] = *(const v8bf*)(arow + c * 64 + kk + 16);   // K = +kA+16 .. +23
#pragma unroll
      for (int j = 0; j < 4; ++j) {
        const bf16_t* srow = sb + (j * 16 + l15) * WPITCH + kk + half * 8;
        union { v16bf v; v8bf h[2]; } b;
        b.h[0] = *(const v8bf*)(srow);
        b.h[1] = *(const v8bf*)(srow + 16);
        acc[j] = __builtin_amdgcn_wmma_f32_16x16x32_bf16(
            false, a.v, false, b.v, (short)0, acc[j], false, false);
      }
    }
    __syncthreads();                       // done reading buf[c&1] before reuse
  }

#pragma unroll
  for (int j = 0; j < 4; ++j) {
    int n = n0 + j * 16 + l15;
    float bv = bias ? bias[n] : 0.0f;
#pragma unroll
    for (int r = 0; r < 8; ++r) {
      int m = tileM * 16 + half * 8 + r;
      if (m < M) C[(size_t)m * ldc + n] = acc[j][r] + bv;
    }
  }
}

// ---------------- elementwise kernels ----------------
__global__ void k_cvt(bf16_t* __restrict__ d, const float* __restrict__ s, int n) {
  int i = blockIdx.x * blockDim.x + threadIdx.x;
  if (i < n) d[i] = f2bf(s[i]);
}
__global__ void k_zero_f(float* p, int n) {
  int i = blockIdx.x * blockDim.x + threadIdx.x;
  if (i < n) p[i] = 0.0f;
}
__global__ void k_zero_bf(bf16_t* p, int n) {
  int i = blockIdx.x * blockDim.x + threadIdx.x;
  if (i < n) p[i] = f2bf(0.0f);
}
__global__ void k_fill_f(float* p, float v, int n) {
  int i = blockIdx.x * blockDim.x + threadIdx.x;
  if (i < n) p[i] = v;
}

// TreeLSTM gates: c,h update. xiou/hiou: [N,3H] (no bias), xf/hf: [N,H].
__global__ void k_tree_gates(float* __restrict__ c, float* __restrict__ h, bf16_t* __restrict__ hb,
                             const float* __restrict__ xiou, const float* __restrict__ hiou,
                             const float* __restrict__ bioux, const float* __restrict__ biouh,
                             const float* __restrict__ xf, const float* __restrict__ hf,
                             const float* __restrict__ bfx, const float* __restrict__ bfh, int cnt) {
  int i = blockIdx.x * blockDim.x + threadIdx.x;
  if (i >= cnt) return;
  int n = i >> 8, hh = i & 255;
  size_t b3 = (size_t)n * G3H, b1 = (size_t)n * HDIM;
  float ii = sigf(xiou[b3 + hh]       + bioux[hh]       + hiou[b3 + hh]       + biouh[hh]);
  float oo = sigf(xiou[b3 + 256 + hh] + bioux[256 + hh] + hiou[b3 + 256 + hh] + biouh[256 + hh]);
  float uu = tanhf(xiou[b3 + 512 + hh] + bioux[512 + hh] + hiou[b3 + 512 + hh] + biouh[512 + hh]);
  float ff = sigf(xf[b1 + hh] + bfx[hh] + hf[b1 + hh] + bfh[hh]);
  float cn = ii * uu + ff * c[i];
  float hn = oo * tanhf(cn);
  c[i] = cn; h[i] = hn; hb[i] = f2bf(hn);
}

// GRU gates; optional extra bf16 output (fv concat slot), stride in elements.
__global__ void k_gru_gates(float* __restrict__ h, bf16_t* __restrict__ hb,
                            const float* __restrict__ xg, const float* __restrict__ gh,
                            const float* __restrict__ bih, const float* __restrict__ bhh,
                            bf16_t* extra, int extra_stride, int cnt) {
  int i = blockIdx.x * blockDim.x + threadIdx.x;
  if (i >= cnt) return;
  int n = i >> 8, hh = i & 255;
  size_t b3 = (size_t)n * G3H;
  float r  = sigf(xg[b3 + hh]       + bih[hh]       + gh[b3 + hh]       + bhh[hh]);
  float z  = sigf(xg[b3 + 256 + hh] + bih[256 + hh] + gh[b3 + 256 + hh] + bhh[256 + hh]);
  float nn = tanhf(xg[b3 + 512 + hh] + bih[512 + hh] + r * (gh[b3 + 512 + hh] + bhh[512 + hh]));
  float hn = (1.0f - z) * nn + z * h[i];
  h[i] = hn; hb[i] = f2bf(hn);
  if (extra) extra[(size_t)n * extra_stride + hh] = f2bf(hn);
}

__global__ void k_store_feat(bf16_t* __restrict__ feats, const float* __restrict__ h, int slot, int cnt) {
  int i = blockIdx.x * blockDim.x + threadIdx.x;
  if (i >= cnt) return;
  int n = i >> 8, hh = i & 255;
  feats[(size_t)n * (5 * HDIM) + slot * HDIM + hh] = f2bf(h[i]);
}

// ---- GCN ----
__global__ void k_deg(float* __restrict__ deg, const int* __restrict__ dst, int e) {
  int i = blockIdx.x * blockDim.x + threadIdx.x;
  if (i < e) atomicAdd(&deg[dst[i]], 1.0f);
}
__global__ void k_rsqrt(float* p, int n) {
  int i = blockIdx.x * blockDim.x + threadIdx.x;
  if (i < n) p[i] = rsqrtf(p[i]);
}
__global__ void k_scatter(float* __restrict__ agg, const float* __restrict__ xw,
                          const int* __restrict__ src, const int* __restrict__ dst,
                          const float* __restrict__ dinv, int e) {
  int ei = blockIdx.x;
  if (ei >= e) return;
  int s = src[ei], d = dst[ei];
  float nrm = dinv[s] * dinv[d];
  int hh = threadIdx.x;
  atomicAdd(&agg[(size_t)d * HDIM + hh], xw[(size_t)s * HDIM + hh] * nrm);
}
__global__ void k_selfloop(float* __restrict__ agg, const float* __restrict__ xw,
                           const float* __restrict__ dinv, int cnt) {
  int i = blockIdx.x * blockDim.x + threadIdx.x;
  if (i >= cnt) return;
  int n = i >> 8;
  float di = dinv[n];
  agg[i] += xw[i] * di * di;
}
__global__ void k_relu_bias(float* __restrict__ xo, bf16_t* __restrict__ xob,
                            const float* __restrict__ agg, const float* __restrict__ b, int cnt) {
  int i = blockIdx.x * blockDim.x + threadIdx.x;
  if (i >= cnt) return;
  float v = agg[i] + b[i & 255];
  v = fmaxf(v, 0.0f);
  xo[i] = v; xob[i] = f2bf(v);
}
// post-ReLU values are >= 0, so int-bit compare == float compare
__global__ void k_maxpool(const float* __restrict__ x, float* __restrict__ out, int cnt) {
  int i = blockIdx.x * blockDim.x + threadIdx.x;
  if (i < cnt) atomicMax((int*)&out[i & 255], __float_as_int(x[i]));
}

// ---------------- host orchestration ----------------
static inline int cdiv_i(int a, int b) { return (a + b - 1) / b; }
#define EL(kern, cnt, ...) kern<<<cdiv_i((int)(cnt), 256), 256, 0, stream>>>(__VA_ARGS__)

extern "C" void kernel_launch(void* const* d_in, const int* in_sizes, int n_in,
                              void* d_out, int out_size, void* d_ws, size_t ws_size,
                              hipStream_t stream) {
  (void)n_in; (void)out_size; (void)ws_size;

  const int Nn = in_sizes[0] / (LDIM * FDIM);   // 10000
  const int Ee = in_sizes[5] / 2;               // 160000

  const float* tree_in = (const float*)d_in[0];
  const float* seqs[4] = {(const float*)d_in[1], (const float*)d_in[2],
                          (const float*)d_in[3], (const float*)d_in[4]};
  const int*   eidx    = (const int*)d_in[5];
  const float* tlWioux = (const float*)d_in[6];
  const float* bioux   = (const float*)d_in[7];
  const float* tlWiouh = (const float*)d_in[8];
  const float* biouh   = (const float*)d_in[9];
  const float* tlWfx   = (const float*)d_in[10];
  const float* bfx     = (const float*)d_in[11];
  const float* tlWfh   = (const float*)d_in[12];
  const float* bfh     = (const float*)d_in[13];
  const float* gruWih  = (const float*)d_in[14];
  const float* gruWhh  = (const float*)d_in[15];
  const float* gru_bih = (const float*)d_in[16];
  const float* gru_bhh = (const float*)d_in[17];
  const float* combWih = (const float*)d_in[18];
  const float* combWhh = (const float*)d_in[19];
  const float* comb_bih= (const float*)d_in[20];
  const float* comb_bhh= (const float*)d_in[21];
  const float* connW   = (const float*)d_in[22];
  const float* connb   = (const float*)d_in[23];
  const float* gcnW    = (const float*)d_in[24];
  const float* gcnb    = (const float*)d_in[25];

  // ---- workspace bump allocator (256B aligned) ----
  uintptr_t cur = (uintptr_t)d_ws;
  auto alloc = [&](size_t bytes) -> void* {
    cur = (cur + 255) & ~(uintptr_t)255;
    void* p = (void*)cur;
    cur += bytes;
    return p;
  };

  // bf16 weights
  bf16_t* w_tlWioux = (bf16_t*)alloc((size_t)G3H * FDIM * 2);
  bf16_t* w_tlWiouh = (bf16_t*)alloc((size_t)G3H * HDIM * 2);
  bf16_t* w_tlWfx   = (bf16_t*)alloc((size_t)HDIM * FDIM * 2);
  bf16_t* w_tlWfh   = (bf16_t*)alloc((size_t)HDIM * HDIM * 2);
  bf16_t* w_gruWih  = (bf16_t*)alloc((size_t)4 * G3H * FDIM * 2);
  bf16_t* w_gruWhh  = (bf16_t*)alloc((size_t)4 * G3H * HDIM * 2);
  bf16_t* w_combWih = (bf16_t*)alloc((size_t)2 * G3H * HDIM * 2);
  bf16_t* w_combWhh = (bf16_t*)alloc((size_t)2 * G3H * HDIM * 2);
  bf16_t* w_connW   = (bf16_t*)alloc((size_t)HDIM * 10 * HDIM * 2);
  bf16_t* w_gcnW    = (bf16_t*)alloc((size_t)3 * HDIM * HDIM * 2);
  // activations
  bf16_t* inbf    = (bf16_t*)alloc((size_t)Nn * TDIM * FDIM * 2);
  bf16_t* featsbf = (bf16_t*)alloc((size_t)Nn * 5 * HDIM * 2);
  bf16_t* fvbf    = (bf16_t*)alloc((size_t)Nn * 10 * HDIM * 2);
  float*  xproj   = (float*)alloc((size_t)Nn * G3H * 4);
  float*  hproj   = (float*)alloc((size_t)Nn * G3H * 4);
  float*  xfp     = (float*)alloc((size_t)Nn * HDIM * 4);
  float*  hfp     = (float*)alloc((size_t)Nn * HDIM * 4);
  float*  cst     = (float*)alloc((size_t)Nn * HDIM * 4);
  float*  hst     = (float*)alloc((size_t)Nn * HDIM * 4);
  bf16_t* hbf     = (bf16_t*)alloc((size_t)Nn * HDIM * 2);
  float*  xcur    = (float*)alloc((size_t)Nn * HDIM * 4);
  bf16_t* xbf     = (bf16_t*)alloc((size_t)Nn * HDIM * 2);
  float*  xw      = (float*)alloc((size_t)Nn * HDIM * 4);
  float*  agg     = (float*)alloc((size_t)Nn * HDIM * 4);
  float*  cvout   = (float*)alloc((size_t)Nn * HDIM * 4);
  float*  dinv    = (float*)alloc((size_t)Nn * 4);

  auto gemm = [&](const bf16_t* A, int lda, const bf16_t* W, int ldw,
                  float* C, int ldc, const float* bias, int M, int Nt, int K) {
    dim3 grid(cdiv_i(M, 128), Nt / 64);
    k_gemm_bf16<<<grid, 256, 0, stream>>>(A, lda, W, ldw, C, ldc, bias, M, Nt, K);
  };

  // ---- convert weights to bf16 ----
  EL(k_cvt, G3H * FDIM,          w_tlWioux, tlWioux, G3H * FDIM);
  EL(k_cvt, G3H * HDIM,          w_tlWiouh, tlWiouh, G3H * HDIM);
  EL(k_cvt, HDIM * FDIM,         w_tlWfx,   tlWfx,   HDIM * FDIM);
  EL(k_cvt, HDIM * HDIM,         w_tlWfh,   tlWfh,   HDIM * HDIM);
  EL(k_cvt, 4 * G3H * FDIM,      w_gruWih,  gruWih,  4 * G3H * FDIM);
  EL(k_cvt, 4 * G3H * HDIM,      w_gruWhh,  gruWhh,  4 * G3H * HDIM);
  EL(k_cvt, 2 * G3H * HDIM,      w_combWih, combWih, 2 * G3H * HDIM);
  EL(k_cvt, 2 * G3H * HDIM,      w_combWhh, combWhh, 2 * G3H * HDIM);
  EL(k_cvt, HDIM * 10 * HDIM,    w_connW,   connW,   HDIM * 10 * HDIM);
  EL(k_cvt, 3 * HDIM * HDIM,     w_gcnW,    gcnW,    3 * HDIM * HDIM);

  const int NH = Nn * HDIM;

  // ---- TreeLSTM over chain (8 steps) ----
  EL(k_cvt, Nn * LDIM * FDIM, inbf, tree_in, Nn * LDIM * FDIM);
  EL(k_zero_f, NH, cst, NH);
  EL(k_zero_f, NH, hst, NH);
  EL(k_zero_bf, NH, hbf, NH);
  for (int t = 0; t < LDIM; ++t) {
    gemm(inbf + (size_t)t * FDIM, LDIM * FDIM, w_tlWioux, FDIM, xproj, G3H, nullptr, Nn, G3H, FDIM);
    gemm(inbf + (size_t)t * FDIM, LDIM * FDIM, w_tlWfx,   FDIM, xfp,   HDIM, nullptr, Nn, HDIM, FDIM);
    gemm(hbf, HDIM, w_tlWiouh, HDIM, hproj, G3H, nullptr, Nn, G3H, HDIM);
    gemm(hbf, HDIM, w_tlWfh,   HDIM, hfp,   HDIM, nullptr, Nn, HDIM, HDIM);
    EL(k_tree_gates, NH, cst, hst, hbf, xproj, hproj, bioux, biouh, xfp, hfp, bfx, bfh, NH);
  }
  EL(k_store_feat, NH, featsbf, hst, 0, NH);

  // ---- 4 GRUs over 16 steps ----
  for (int g = 0; g < 4; ++g) {
    EL(k_cvt, Nn * TDIM * FDIM, inbf, seqs[g], Nn * TDIM * FDIM);
    EL(k_zero_f, NH, hst, NH);
    EL(k_zero_bf, NH, hbf, NH);
    for (int t = 0; t < TDIM; ++t) {
      gemm(inbf + (size_t)t * FDIM, TDIM * FDIM, w_gruWih + (size_t)g * G3H * FDIM, FDIM,
           xproj, G3H, nullptr, Nn, G3H, FDIM);
      gemm(hbf, HDIM, w_gruWhh + (size_t)g * G3H * HDIM, HDIM, hproj, G3H, nullptr, Nn, G3H, HDIM);
      EL(k_gru_gates, NH, hst, hbf, xproj, hproj, gru_bih + g * G3H, gru_bhh + g * G3H,
         (bf16_t*)nullptr, 0, NH);
    }
    EL(k_store_feat, NH, featsbf, hst, g + 1, NH);
  }

  // ---- bidirectional combine GRU over 5 steps, writes fv concat ----
  for (int dir = 0; dir < 2; ++dir) {
    EL(k_zero_f, NH, hst, NH);
    EL(k_zero_bf, NH, hbf, NH);
    for (int s = 0; s < 5; ++s) {
      int t = (dir == 0) ? s : (4 - s);
      gemm(featsbf + (size_t)t * HDIM, 5 * HDIM, w_combWih + (size_t)dir * G3H * HDIM, HDIM,
           xproj, G3H, nullptr, Nn, G3H, HDIM);
      gemm(hbf, HDIM, w_combWhh + (size_t)dir * G3H * HDIM, HDIM, hproj, G3H, nullptr, Nn, G3H, HDIM);
      EL(k_gru_gates, NH, hst, hbf, xproj, hproj, comb_bih + dir * G3H, comb_bhh + dir * G3H,
         fvbf + (size_t)t * 2 * HDIM + (size_t)dir * HDIM, 10 * HDIM, NH);
    }
  }

  // ---- connect: fv[N,2560] @ W[256,2560]^T + b ----
  gemm(fvbf, 10 * HDIM, w_connW, 10 * HDIM, xcur, HDIM, connb, Nn, HDIM, 10 * HDIM);
  EL(k_cvt, NH, xbf, xcur, NH);

  // ---- GCN normalization (self-loops -> deg starts at 1) ----
  EL(k_fill_f, Nn, dinv, 1.0f, Nn);
  EL(k_deg, Ee, dinv, eidx + Ee, Ee);
  EL(k_rsqrt, Nn, dinv, Nn);

  auto gcn_apply = [&](int i, float* outb) {
    gemm(xbf, HDIM, w_gcnW + (size_t)i * HDIM * HDIM, HDIM, xw, HDIM, nullptr, Nn, HDIM, HDIM);
    EL(k_zero_f, NH, agg, NH);
    k_scatter<<<Ee, 256, 0, stream>>>(agg, xw, eidx, eidx + Ee, dinv, Ee);
    EL(k_selfloop, NH, agg, xw, dinv, NH);
    EL(k_relu_bias, NH, outb, xbf, agg, gcnb + i * HDIM, NH);
  };
  for (int i = 0; i < 3; ++i) gcn_apply(i, xcur);
  gcn_apply(2, cvout);   // last conv applied a second time for conv_output

  // ---- global max pool -> d_out [1,256] ----
  EL(k_zero_f, HDIM, (float*)d_out, HDIM);
  EL(k_maxpool, NH, cvout, (float*)d_out, NH);
}
